// PVManifoldMLR_61443802137151
// MI455X (gfx1250) — matrix operand order repaired and checked by reading, not compiled, not run on hardware
//
#include <hip/hip_runtime.h>
#include <math.h>

typedef _Float16 v4h  __attribute__((ext_vector_type(4)));
typedef _Float16 v8h  __attribute__((ext_vector_type(8)));
typedef _Float16 v16h __attribute__((ext_vector_type(16)));
typedef float    v4f  __attribute__((ext_vector_type(4)));
typedef float    v8f  __attribute__((ext_vector_type(8)));

#define B_ROWS 8192
#define DIM    128
#define KCLS   32000
#define LN2F   0.69314718055994530942f

// ---------------------------------------------------------------------------
// Per-class prep: z[K,128] fp32 -> zh f16; c1 = cosh(sr)/||z||,
// c2 = -sinh(sr)  (negated so epilogue is a single fma),
// c3 = ||z|| * ln2 (ln2 folded in: asinh computed via v_log_f32 = log2).
// One wave (32 lanes) per class, 4 floats per lane.
// ---------------------------------------------------------------------------
__global__ void __launch_bounds__(256) prep_z_kernel(
    const float* __restrict__ z, const float* __restrict__ r,
    _Float16* __restrict__ zh, float* __restrict__ c1,
    float* __restrict__ c2, float* __restrict__ c3) {
  const int wave = threadIdx.x >> 5;
  const int lane = threadIdx.x & 31;
  const int cls  = blockIdx.x * 8 + wave;

  const float* row = z + (size_t)cls * DIM + lane * 4;
  v4f v = *(const v4f*)row;

  v4h h;
  h.x = (_Float16)v.x; h.y = (_Float16)v.y;
  h.z = (_Float16)v.z; h.w = (_Float16)v.w;
  *(v4h*)(zh + (size_t)cls * DIM + lane * 4) = h;

  float s = v.x * v.x + v.y * v.y + v.z * v.z + v.w * v.w;
  #pragma unroll
  for (int m = 16; m >= 1; m >>= 1) s += __shfl_xor(s, m, 32);

  if (lane == 0) {
    float zn = fmaxf(sqrtf(s), 1e-15f);
    float sr = fminf(fmaxf(r[cls], -15.0f), 15.0f);  // SQRT_NEG_K == 1
    c1[cls] = coshf(sr) / zn;
    c2[cls] = -sinhf(sr);
    c3[cls] = zn * LN2F;
  }
}

// ---------------------------------------------------------------------------
// Per-row prep: x[B,128] fp32 -> xh f16; beta = sqrt(1 + ||x||^2).
// ---------------------------------------------------------------------------
__global__ void __launch_bounds__(256) prep_x_kernel(
    const float* __restrict__ x, _Float16* __restrict__ xh,
    float* __restrict__ beta) {
  const int wave = threadIdx.x >> 5;
  const int lane = threadIdx.x & 31;
  const int row  = blockIdx.x * 8 + wave;

  const float* p = x + (size_t)row * DIM + lane * 4;
  v4f v = *(const v4f*)p;

  v4h h;
  h.x = (_Float16)v.x; h.y = (_Float16)v.y;
  h.z = (_Float16)v.z; h.w = (_Float16)v.w;
  *(v4h*)(xh + (size_t)row * DIM + lane * 4) = h;

  float s = v.x * v.x + v.y * v.y + v.z * v.z + v.w * v.w;
  #pragma unroll
  for (int m = 16; m >= 1; m >>= 1) s += __shfl_xor(s, m, 32);

  if (lane == 0) beta[row] = sqrtf(1.0f + s);
}

// ---------------------------------------------------------------------------
// Main GEMM + hyperbolic epilogue.
// Block: 256 threads = 8 waves; block tile = 32(M) x 256(N).
// Wave tile: 16(M) x 64(N) = 4 WMMA accumulators, K=128 in 4 steps of 32.
// Epilogue: out = c3*log2(|a| + sqrt(a^2+1)) with sign of a,
//           a = med3(c1*xz - sinh*beta, -1e6, 1e6).
// ---------------------------------------------------------------------------
__global__ void __launch_bounds__(256) gemm_epilogue_kernel(
    const _Float16* __restrict__ xh, const _Float16* __restrict__ zh,
    const float* __restrict__ c1, const float* __restrict__ c2,
    const float* __restrict__ c3, const float* __restrict__ beta,
    float* __restrict__ out) {
  const int lane = threadIdx.x & 31;
  const int wave = threadIdx.x >> 5;
  const int lo   = lane & 15;
  const int hi   = lane >> 4;

  const int m0 = blockIdx.y * 32 + (wave >> 2) * 16;   // wave's M origin
  const int n0 = blockIdx.x * 256 + (wave & 3) * 64;   // wave's N origin

  v8f acc[4];
  #pragma unroll
  for (int j = 0; j < 4; ++j)
    #pragma unroll
    for (int e = 0; e < 8; ++e) acc[j][e] = 0.0f;

  // A fragment addressing (ISA 7.12.2, 16-bit A 16x32):
  //   lanes 0-15 : M=lane, VGPR0-3 = K 0..7,  VGPR4-7 = K 16..23
  //   lanes 16-31: M=lane-16,       K 8..15,            K 24..31
  const _Float16* aBase = xh + (size_t)(m0 + lo) * DIM + hi * 8;

  #pragma unroll
  for (int kk = 0; kk < DIM; kk += 32) {
    v8h a0 = *(const v8h*)(aBase + kk);
    v8h a1 = *(const v8h*)(aBase + kk + 16);
    v16h a = __builtin_shufflevector(a0, a1, 0, 1, 2, 3, 4, 5, 6, 7,
                                     8, 9, 10, 11, 12, 13, 14, 15);
    #pragma unroll
    for (int j = 0; j < 4; ++j) {
      // B fragment (32x16): lanes 0-15: N=lane, K 0..7, 16..23;
      //                     lanes 16-31: N=lane-16, K 8..15, 24..31.
      const _Float16* bp =
          zh + (size_t)(n0 + j * 16 + lo) * DIM + hi * 8 + kk;
      v8h b0 = *(const v8h*)bp;
      v8h b1 = *(const v8h*)(bp + 16);
      v16h b = __builtin_shufflevector(b0, b1, 0, 1, 2, 3, 4, 5, 6, 7,
                                       8, 9, 10, 11, 12, 13, 14, 15);
      acc[j] = __builtin_amdgcn_wmma_f32_16x16x32_f16(
          false, a, false, b, (short)0, acc[j], false, false);
    }
  }

  // Hoist the 8 beta values (one per C VGPR row) out of the j-loop.
  float bv[8];
  #pragma unroll
  for (int g = 0; g < 8; ++g) bv[g] = beta[m0 + hi * 8 + g];

  // Epilogue: C layout (32-bit 16x16): VGPR g -> M = g + 8*hi, N = lo.
  #pragma unroll
  for (int j = 0; j < 4; ++j) {
    const int n = n0 + j * 16 + lo;
    const float k1 = c1[n];   // cosh/||z||
    const float k2 = c2[n];   // -sinh
    const float k3 = c3[n];   // ||z||*ln2
    #pragma unroll
    for (int g = 0; g < 8; ++g) {
      const int m = m0 + hi * 8 + g;
      float a = __builtin_fmaf(k2, bv[g], k1 * acc[j][g]);
      a = __builtin_amdgcn_fmed3f(a, -1.0e6f, 1.0e6f);      // clip
      const float t  = __builtin_fabsf(a);
      const float sq = __builtin_amdgcn_sqrtf(__builtin_fmaf(t, t, 1.0f));
      const float l2 = __builtin_amdgcn_logf(t + sq);        // log2(s), s>=1
      out[(size_t)m * KCLS + n] = __builtin_copysignf(l2 * k3, a);
    }
  }
}

// ---------------------------------------------------------------------------
extern "C" void kernel_launch(void* const* d_in, const int* in_sizes, int n_in,
                              void* d_out, int out_size, void* d_ws, size_t ws_size,
                              hipStream_t stream) {
  (void)in_sizes; (void)n_in; (void)out_size; (void)ws_size;
  const float* x = (const float*)d_in[0];   // [8192,128]
  const float* z = (const float*)d_in[1];   // [32000,128]
  const float* r = (const float*)d_in[2];   // [32000,1]
  float* out = (float*)d_out;               // [8192,32000]

  // Workspace layout (≈10.7 MB)
  char* ws = (char*)d_ws;
  constexpr size_t XH_BYTES = (size_t)B_ROWS * DIM * 2;   // 2,097,152
  constexpr size_t ZH_BYTES = (size_t)KCLS * DIM * 2;     // 8,192,000
  constexpr size_t C_BYTES  = (size_t)KCLS * 4;           //   128,000
  _Float16* xh  = (_Float16*)(ws);
  _Float16* zh  = (_Float16*)(ws + XH_BYTES);
  float*    c1  = (float*)(ws + XH_BYTES + ZH_BYTES);
  float*    c2  = (float*)(ws + XH_BYTES + ZH_BYTES + C_BYTES);
  float*    c3  = (float*)(ws + XH_BYTES + ZH_BYTES + 2 * C_BYTES);
  float*    bta = (float*)(ws + XH_BYTES + ZH_BYTES + 3 * C_BYTES);

  prep_z_kernel<<<KCLS / 8, 256, 0, stream>>>(z, r, zh, c1, c2, c3);
  prep_x_kernel<<<B_ROWS / 8, 256, 0, stream>>>(x, xh, bta);

  dim3 grid(KCLS / 256, B_ROWS / 32);  // (125, 256)
  gemm_epilogue_kernel<<<grid, 256, 0, stream>>>(xh, zh, c1, c2, c3, bta, out);
}
